// MainModalityDeformableMoE_79551384257140
// MI455X (gfx1250) — matrix operand order repaired and checked by reading, not compiled, not run on hardware
//
#include <hip/hip_runtime.h>

// ---------------------------------------------------------------------------
// MainModalityDeformableMoE for MI455X (gfx1250, wave32, WMMA)
// Deformable-attention collapses exactly (Lq=Lk=1): attn = repeat(src, NH),
// so attn@Wproj == src @ (sum of Wproj head blocks). Woff/Watt are dead.
// GEMMs: bf16 WMMA (v_wmma_f32_16x16x32_bf16), 256x128 block tile staged in
// LDS with double-buffered GLOBAL_LOAD_ASYNC_TO_LDS_B128 (ASYNCcnt) when the
// builtin exists, else sync b128 loads + LDS stores. 64x64 wave tile.
// ---------------------------------------------------------------------------

#define Hdim  1024
#define NHEAD 4
#define PRED  20
#define FFdim 4096
#define Bsz   8192
#define F0    2048
#define F1    1024
#define F2    512

typedef __attribute__((ext_vector_type(16))) __bf16 bf16x16;
typedef __attribute__((ext_vector_type(8)))  float  f32x8;
typedef int v4i __attribute__((vector_size(16)));   // matches builtin param

#define AS1 __attribute__((address_space(1)))
#define AS3 __attribute__((address_space(3)))

#if defined(__has_builtin)
#  if __has_builtin(__builtin_amdgcn_global_load_async_to_lds_b128)
#    define USE_ASYNC_LDS 1
#  endif
#endif
#ifndef USE_ASYNC_LDS
#  define USE_ASYNC_LDS 0
#endif

static __device__ __forceinline__ void async_copy_b128(const void* g, void* l) {
#if USE_ASYNC_LDS
  // (global src, LDS dst, imm offset, imm cpol); AS-qualified typed pointers
  // convert implicitly to generic if the prototype is unqualified.
  __builtin_amdgcn_global_load_async_to_lds_b128((AS1 v4i*)g, (AS3 v4i*)l,
                                                 0, 0);
#else
  *(uint4*)l = *(const uint4*)g;
#endif
}

static __device__ __forceinline__ void wait_async() {
#if USE_ASYNC_LDS
#  if __has_builtin(__builtin_amdgcn_s_wait_asynccnt)
  __builtin_amdgcn_s_wait_asynccnt(0);
#  else
  asm volatile("s_wait_asynccnt 0x0" ::: "memory");
#  endif
#endif
}

static __device__ __forceinline__ unsigned short f2bf(float x) {
  union { float f; unsigned u; } v; v.f = x;
  unsigned r = v.u + 0x7FFFu + ((v.u >> 16) & 1u);   // round-to-nearest-even
  return (unsigned short)(r >> 16);
}

// ---------------------------------------------------------------------------
// GEMM: C(MxN,f32) = A(MxK,bf16) @ Bt(NxK,bf16)^T + bias, epilogue activation.
// Block: 256 threads (8 waves); block tile 256x128; wave tile 64x64 (4x4 WMMA
// accumulators). K staged through double-buffered LDS in BK=32 slices.
// ACT: 0=none, 1=relu, 2=gelu(exact erf)
// ---------------------------------------------------------------------------
#define BM   256
#define BN   128
#define BK   32
#define KPAD 40   // +8 bf16 pad -> 80B row stride, conflict-free b128 reads

template <int ACT, bool WF32, bool WB16>
__global__ __launch_bounds__(256) void gemm_bf16_kernel(
    const unsigned short* __restrict__ A,
    const unsigned short* __restrict__ Bt,
    const float* __restrict__ bias,
    float* __restrict__ Cf,
    unsigned short* __restrict__ Cb,
    int M, int N, int K)
{
  __shared__ __align__(16) unsigned short sA[2][BM][KPAD];
  __shared__ __align__(16) unsigned short sB[2][BN][KPAD];

  const int t    = threadIdx.x;
  const int lane = t & 31;
  const int wave = t >> 5;
  const int blockRow = blockIdx.y * BM;
  const int blockCol = blockIdx.x * BN;
  const int mw = (wave & 3) * 64;     // wave row offset inside block tile
  const int nw = (wave >> 2) * 64;    // wave col offset inside block tile

  f32x8 acc[4][4];
  for (int i = 0; i < 4; ++i)
    for (int j = 0; j < 4; ++j)
      for (int r = 0; r < 8; ++r) acc[i][j][r] = 0.0f;

  const int steps = K / BK;

  auto stage = [&](int buf, int k0) {
    // A tile: BM x BK bf16 = 1024 x 16B chunks -> 4 per thread
    for (int q = 0; q < 4; ++q) {
      int ch  = t + 256 * q;
      int row = ch >> 2;
      int col = (ch & 3) * 8;
      async_copy_b128(A + (size_t)(blockRow + row) * K + k0 + col,
                      &sA[buf][row][col]);
    }
    // B tile: BN x BK bf16 = 512 x 16B chunks -> 2 per thread
    for (int q = 0; q < 2; ++q) {
      int ch  = t + 256 * q;
      int row = ch >> 2;
      int col = (ch & 3) * 8;
      async_copy_b128(Bt + (size_t)(blockCol + row) * K + k0 + col,
                      &sB[buf][row][col]);
    }
  };

  stage(0, 0);
  for (int s = 0; s < steps; ++s) {
    const int cur = s & 1;
    wait_async();        // our stage-s copies are in LDS
    __syncthreads();     // everyone's stage-s copies in LDS; prior reads done
    if (s + 1 < steps) stage(cur ^ 1, (s + 1) * BK);

    // fragment loads from LDS (ds_load_b128)
    bf16x16 af[4], bfm[4];
    {
      const int m0 = mw + (lane & 15);
      const int ka = (lane >> 4) << 3;   // 0 or 8
      for (int i = 0; i < 4; ++i) {
        const unsigned short* p = &sA[cur][m0 + 16 * i][ka];
        union { uint4 u[2]; bf16x16 v; } r;
        r.u[0] = *(const uint4*)(p);       // K = ka .. ka+7
        r.u[1] = *(const uint4*)(p + 16);  // K = ka+16 .. ka+23
        af[i] = r.v;
      }
      const int n0 = nw + (lane & 15);
      const int kb = (lane >> 4) << 3;   // 0 or 8
      for (int j = 0; j < 4; ++j) {
        const unsigned short* p = &sB[cur][n0 + 16 * j][kb];
        union { uint4 u[2]; bf16x16 v; } r;
        r.u[0] = *(const uint4*)(p);       // K = kb .. kb+7
        r.u[1] = *(const uint4*)(p + 16);  // K = kb+16 .. kb+23
        bfm[j] = r.v;
      }
    }
    for (int i = 0; i < 4; ++i)
      for (int j = 0; j < 4; ++j)
        acc[i][j] = __builtin_amdgcn_wmma_f32_16x16x32_bf16(
            false, af[i], false, bfm[j], (short)0, acc[i][j], false, false);
  }

  // C/D layout: VGPR r -> M = r (lanes 0-15) or 8+r (lanes 16-31), N = lane&15
  for (int i = 0; i < 4; ++i) {
    int mbase = blockRow + mw + 16 * i + ((lane >> 4) << 3);
    for (int j = 0; j < 4; ++j) {
      int n = blockCol + nw + 16 * j + (lane & 15);
      float bv = bias[n];
      for (int r = 0; r < 8; ++r) {
        float v = acc[i][j][r] + bv;
        if (ACT == 1) v = fmaxf(v, 0.0f);
        if (ACT == 2) v = 0.5f * v * (1.0f + erff(v * 0.70710678118654752f));
        size_t o = (size_t)(mbase + r) * N + n;
        if (WF32) Cf[o] = v;
        if (WB16) Cb[o] = f2bf(v);
      }
    }
  }
}

// ---------------------------------------------------------------------------
// Elementwise / prep kernels
// ---------------------------------------------------------------------------
__global__ void f32_to_bf16_kernel(const float* __restrict__ in,
                                   unsigned short* __restrict__ out, size_t n) {
  size_t i = (size_t)blockIdx.x * blockDim.x + threadIdx.x;
  if (i < n) out[i] = f2bf(in[i]);
}

__global__ void xm1_mean_kernel(const float* __restrict__ x,
                                unsigned short* __restrict__ out) {
  size_t i = (size_t)blockIdx.x * blockDim.x + threadIdx.x;  // b*H + h
  if (i >= (size_t)Bsz * Hdim) return;
  size_t b = i >> 10; int h = (int)(i & 1023);
  const float* p = x + (b * 8) * (size_t)Hdim + h;
  float s = 0.f;
  for (int k = 0; k < 8; ++k) s += p[(size_t)k * Hdim];
  out[i] = f2bf(s * 0.125f);
}

// W[K][N] f32  ->  Wt[N][K] bf16
__global__ void conv_transpose_kernel(const float* __restrict__ W,
                                      unsigned short* __restrict__ Wt,
                                      int K, int N) {
  size_t i = (size_t)blockIdx.x * blockDim.x + threadIdx.x;
  if (i >= (size_t)K * N) return;
  int k = (int)(i / N), n = (int)(i % N);
  Wt[(size_t)n * K + k] = f2bf(W[i]);
}

// Wsum_t[n][k] = bf16( sum_h Wproj[h*H + k][n] )  (attn == repeat(src, NH))
__global__ void wproj_sum_transpose_kernel(const float* __restrict__ Wproj,
                                           unsigned short* __restrict__ Wt) {
  size_t i = (size_t)blockIdx.x * blockDim.x + threadIdx.x;  // k*H + n
  if (i >= (size_t)Hdim * Hdim) return;
  int k = (int)(i >> 10), n = (int)(i & 1023);
  float s = 0.f;
  for (int hh = 0; hh < NHEAD; ++hh)
    s += Wproj[((size_t)hh * Hdim + k) * Hdim + n];
  Wt[(size_t)n * Hdim + k] = f2bf(s);
}

// ---------------------------------------------------------------------------
// Gate: softmax(cat@Wg+bg) then fused = sum_m gw[m]*r_m.  One wave per row.
// ---------------------------------------------------------------------------
__global__ __launch_bounds__(256) void gate_fuse_kernel(
    const float* __restrict__ r0, const float* __restrict__ r1,
    const float* __restrict__ r2, const float* __restrict__ Wg,
    const float* __restrict__ bg,
    float* __restrict__ fusedF, unsigned short* __restrict__ fusedB)
{
  int lane = threadIdx.x & 31;
  int b = blockIdx.x * 8 + (threadIdx.x >> 5);
  const float* p0 = r0 + (size_t)b * Hdim;
  const float* p1 = r1 + (size_t)b * Hdim;
  const float* p2 = r2 + (size_t)b * Hdim;

  float s0 = 0.f, s1 = 0.f, s2 = 0.f;
  for (int h = lane; h < Hdim; h += 32) {
    float v0 = p0[h], v1 = p1[h], v2 = p2[h];
    const float* w0 = Wg + (size_t)h * 3;
    const float* w1 = Wg + (size_t)(Hdim + h) * 3;
    const float* w2 = Wg + (size_t)(2 * Hdim + h) * 3;
    s0 += v0 * w0[0] + v1 * w1[0] + v2 * w2[0];
    s1 += v0 * w0[1] + v1 * w1[1] + v2 * w2[1];
    s2 += v0 * w0[2] + v1 * w1[2] + v2 * w2[2];
  }
  for (int m = 16; m >= 1; m >>= 1) {
    s0 += __shfl_xor(s0, m, 32);
    s1 += __shfl_xor(s1, m, 32);
    s2 += __shfl_xor(s2, m, 32);
  }
  s0 += bg[0]; s1 += bg[1]; s2 += bg[2];
  float mx = fmaxf(s0, fmaxf(s1, s2));
  float e0 = expf(s0 - mx), e1 = expf(s1 - mx), e2 = expf(s2 - mx);
  float inv = 1.0f / (e0 + e1 + e2);
  float g0 = e0 * inv, g1 = e1 * inv, g2 = e2 * inv;

  for (int h = lane; h < Hdim; h += 32) {
    float f = g0 * p0[h] + g1 * p1[h] + g2 * p2[h];
    fusedF[(size_t)b * Hdim + h] = f;
    fusedB[(size_t)b * Hdim + h] = f2bf(f);
  }
}

// ---------------------------------------------------------------------------
// out = LayerNorm(X + Y) * g + be ; one 256-thread block per row (H=1024)
// ---------------------------------------------------------------------------
__global__ __launch_bounds__(256) void add_ln_kernel(
    const float* __restrict__ X, const float* __restrict__ Y,
    const float* __restrict__ g, const float* __restrict__ be,
    float* __restrict__ outF, unsigned short* __restrict__ outB)
{
  __shared__ float sSum[8], sSq[8];
  int b = blockIdx.x;
  int t = threadIdx.x;
  int lane = t & 31, wave = t >> 5;
  const float* x = X + (size_t)b * Hdim;
  const float* y = Y + (size_t)b * Hdim;

  float v[4]; float sum = 0.f, sq = 0.f;
  for (int c = 0; c < 4; ++c) {
    int h = t + 256 * c;
    float z = x[h] + y[h];
    v[c] = z; sum += z; sq += z * z;
  }
  for (int m = 16; m >= 1; m >>= 1) {
    sum += __shfl_xor(sum, m, 32);
    sq  += __shfl_xor(sq,  m, 32);
  }
  if (lane == 0) { sSum[wave] = sum; sSq[wave] = sq; }
  __syncthreads();
  float tot = 0.f, totq = 0.f;
  for (int w = 0; w < 8; ++w) { tot += sSum[w]; totq += sSq[w]; }
  float mu   = tot * (1.0f / Hdim);
  float var  = totq * (1.0f / Hdim) - mu * mu;
  float rstd = rsqrtf(var + 1e-5f);
  for (int c = 0; c < 4; ++c) {
    int h = t + 256 * c;
    float o = (v[c] - mu) * rstd * g[h] + be[h];
    outF[(size_t)b * Hdim + h] = o;
    if (outB) outB[(size_t)b * Hdim + h] = f2bf(o);
  }
}

// ---------------------------------------------------------------------------
// Head: logits = src@Wh + bh; hazards = sigmoid; survival = cumprod(1-h).
// One wave per row.
// ---------------------------------------------------------------------------
__global__ __launch_bounds__(256) void head_kernel(
    const float* __restrict__ S, const float* __restrict__ Wh,
    const float* __restrict__ bh,
    float* __restrict__ outHaz, float* __restrict__ outSurv)
{
  int lane = threadIdx.x & 31;
  int b = blockIdx.x * 8 + (threadIdx.x >> 5);
  const float* s = S + (size_t)b * Hdim;

  float acc[PRED];
  for (int p = 0; p < PRED; ++p) acc[p] = 0.f;
  for (int h = lane; h < Hdim; h += 32) {
    float v = s[h];
    const float* w = Wh + (size_t)h * PRED;
    for (int p = 0; p < PRED; ++p) acc[p] += v * w[p];
  }
  for (int p = 0; p < PRED; ++p)
    for (int m = 16; m >= 1; m >>= 1) acc[p] += __shfl_xor(acc[p], m, 32);

  if (lane == 0) {
    float surv = 1.0f;
    for (int p = 0; p < PRED; ++p) {
      float z  = acc[p] + bh[p];
      float hz = 1.0f / (1.0f + expf(-z));
      outHaz[(size_t)b * PRED + p] = hz;
      surv *= (1.0f - hz);
      outSurv[(size_t)b * PRED + p] = surv;
    }
  }
}

// ---------------------------------------------------------------------------
// Host-side orchestration
// ---------------------------------------------------------------------------
extern "C" void kernel_launch(void* const* d_in, const int* in_sizes, int n_in,
                              void* d_out, int out_size, void* d_ws, size_t ws_size,
                              hipStream_t stream) {
  const float* x_m0  = (const float*)d_in[0];
  const float* x_m1  = (const float*)d_in[1];
  const float* x_m2  = (const float*)d_in[2];
  const float* Wp_m0 = (const float*)d_in[3];
  const float* bp_m0 = (const float*)d_in[4];
  const float* Wp_m1 = (const float*)d_in[5];
  const float* bp_m1 = (const float*)d_in[6];
  const float* Wp_m2 = (const float*)d_in[7];
  const float* bp_m2 = (const float*)d_in[8];
  const float* Wg    = (const float*)d_in[9];
  const float* bg    = (const float*)d_in[10];
  // d_in[11..14] = Woff/boff/Watt/batt: mathematically dead (Lk==1)
  const float* Wproj = (const float*)d_in[15];
  const float* bproj = (const float*)d_in[16];
  const float* W1    = (const float*)d_in[17];
  const float* b1    = (const float*)d_in[18];
  const float* W2    = (const float*)d_in[19];
  const float* b2    = (const float*)d_in[20];
  const float* g1    = (const float*)d_in[21];
  const float* be1   = (const float*)d_in[22];
  const float* g2    = (const float*)d_in[23];
  const float* be2   = (const float*)d_in[24];
  const float* Wh    = (const float*)d_in[25];
  const float* bh    = (const float*)d_in[26];

  char* ws = (char*)d_ws;
  // --- workspace layout (bytes), sequential with reuse ---
  const size_t o_xm0b  = 0;                          // 8192x2048 bf16 = 32MB
  const size_t o_xm1b  = o_xm0b  + (size_t)Bsz * F0 * 2;      // 16MB
  const size_t o_xm2b  = o_xm1b  + (size_t)Bsz * F1 * 2;      // 8MB
  const size_t o_wp0t  = o_xm2b  + (size_t)Bsz * F2 * 2;      // 4MB
  const size_t o_wp1t  = o_wp0t  + (size_t)F0 * Hdim * 2;     // 2MB
  const size_t o_wp2t  = o_wp1t  + (size_t)F1 * Hdim * 2;     // 1MB
  const size_t o_wsumt = o_wp2t  + (size_t)F2 * Hdim * 2;     // 2MB
  const size_t o_w1t   = o_wsumt + (size_t)Hdim * Hdim * 2;   // 8MB
  const size_t o_w2t   = o_w1t   + (size_t)Hdim * FFdim * 2;  // 8MB
  const size_t o_r0    = o_w2t   + (size_t)FFdim * Hdim * 2;  // 32MB
  const size_t o_r1    = o_r0    + (size_t)Bsz * Hdim * 4;    // 32MB
  const size_t o_r2    = o_r1    + (size_t)Bsz * Hdim * 4;    // 32MB
  const size_t o_fusedf= o_r2    + (size_t)Bsz * Hdim * 4;    // 32MB
  const size_t o_fusedb= o_fusedf+ (size_t)Bsz * Hdim * 4;    // 16MB
  const size_t o_h1b   = o_fusedb+ (size_t)Bsz * Hdim * 2;    // 64MB
  const size_t o_srcb  = o_h1b   + (size_t)Bsz * FFdim * 2;   // 32MB
  // reuse (lifetimes are disjoint):
  const size_t o_proj  = o_r0;      // attn-proj output (r0 dead after fusion)
  const size_t o_srcaf = o_r1;      // LN1 f32 output
  const size_t o_srcab = o_r2;      // LN1 bf16 output (16MB into 32MB slot)
  const size_t o_h2    = o_fusedf;  // FFN2 output (fused dead after LN1)

  unsigned short* xm0b   = (unsigned short*)(ws + o_xm0b);
  unsigned short* xm1b   = (unsigned short*)(ws + o_xm1b);
  unsigned short* xm2b   = (unsigned short*)(ws + o_xm2b);
  unsigned short* wp0t   = (unsigned short*)(ws + o_wp0t);
  unsigned short* wp1t   = (unsigned short*)(ws + o_wp1t);
  unsigned short* wp2t   = (unsigned short*)(ws + o_wp2t);
  unsigned short* wsumt  = (unsigned short*)(ws + o_wsumt);
  unsigned short* w1t    = (unsigned short*)(ws + o_w1t);
  unsigned short* w2t    = (unsigned short*)(ws + o_w2t);
  float*          r0     = (float*)(ws + o_r0);
  float*          r1     = (float*)(ws + o_r1);
  float*          r2     = (float*)(ws + o_r2);
  float*          fusedf = (float*)(ws + o_fusedf);
  unsigned short* fusedb = (unsigned short*)(ws + o_fusedb);
  unsigned short* h1b    = (unsigned short*)(ws + o_h1b);
  float*          srcbf  = (float*)(ws + o_srcb);
  float*          proj   = (float*)(ws + o_proj);
  float*          srcaf  = (float*)(ws + o_srcaf);
  unsigned short* srcab  = (unsigned short*)(ws + o_srcab);
  float*          h2     = (float*)(ws + o_h2);

  float* outHaz  = (float*)d_out;
  float* outSurv = outHaz + (size_t)Bsz * PRED;

  const int T = 256;
  auto blocks = [](size_t n) { return (unsigned)((n + 255) / 256); };

  // ---- input / weight preparation ----
  f32_to_bf16_kernel<<<blocks((size_t)Bsz * F0), T, 0, stream>>>(x_m0, xm0b, (size_t)Bsz * F0);
  xm1_mean_kernel   <<<blocks((size_t)Bsz * Hdim), T, 0, stream>>>(x_m1, xm1b);
  f32_to_bf16_kernel<<<blocks((size_t)Bsz * F2), T, 0, stream>>>(x_m2, xm2b, (size_t)Bsz * F2);
  conv_transpose_kernel<<<blocks((size_t)F0 * Hdim), T, 0, stream>>>(Wp_m0, wp0t, F0, Hdim);
  conv_transpose_kernel<<<blocks((size_t)F1 * Hdim), T, 0, stream>>>(Wp_m1, wp1t, F1, Hdim);
  conv_transpose_kernel<<<blocks((size_t)F2 * Hdim), T, 0, stream>>>(Wp_m2, wp2t, F2, Hdim);
  wproj_sum_transpose_kernel<<<blocks((size_t)Hdim * Hdim), T, 0, stream>>>(Wproj, wsumt);
  conv_transpose_kernel<<<blocks((size_t)Hdim * FFdim), T, 0, stream>>>(W1, w1t, Hdim, FFdim);
  conv_transpose_kernel<<<blocks((size_t)FFdim * Hdim), T, 0, stream>>>(W2, w2t, FFdim, Hdim);

  // ---- GEMM grids: (N/BN, M/BM) ----
  dim3 gN1024(Hdim / BN, Bsz / BM);    // (8, 32)
  dim3 gN4096(FFdim / BN, Bsz / BM);   // (32, 32)

  // ---- modality projections (relu) ----
  gemm_bf16_kernel<1, true, false><<<gN1024, T, 0, stream>>>(xm0b, wp0t, bp_m0, r0, nullptr, Bsz, Hdim, F0);
  gemm_bf16_kernel<1, true, false><<<gN1024, T, 0, stream>>>(xm1b, wp1t, bp_m1, r1, nullptr, Bsz, Hdim, F1);
  gemm_bf16_kernel<1, true, false><<<gN1024, T, 0, stream>>>(xm2b, wp2t, bp_m2, r2, nullptr, Bsz, Hdim, F2);

  // ---- gating softmax + fusion ----
  gate_fuse_kernel<<<Bsz / 8, T, 0, stream>>>(r0, r1, r2, Wg, bg, fusedf, fusedb);

  // ---- collapsed deformable attention projection ----
  gemm_bf16_kernel<0, true, false><<<gN1024, T, 0, stream>>>(fusedb, wsumt, bproj, proj, nullptr, Bsz, Hdim, Hdim);
  add_ln_kernel<<<Bsz, T, 0, stream>>>(fusedf, proj, g1, be1, srcaf, srcab);

  // ---- FFN (gelu) ----
  gemm_bf16_kernel<2, false, true><<<gN4096, T, 0, stream>>>(srcab, w1t, b1, nullptr, h1b, Bsz, FFdim, Hdim);
  gemm_bf16_kernel<0, true, false><<<gN1024, T, 0, stream>>>(h1b, w2t, b2, h2, nullptr, Bsz, Hdim, FFdim);
  add_ln_kernel<<<Bsz, T, 0, stream>>>(srcaf, h2, g2, be2, srcbf, nullptr);

  // ---- survival head ----
  head_kernel<<<Bsz / 8, T, 0, stream>>>(srcbf, Wh, bh, outHaz, outSurv);
}
